// GraphContourLabeller_23493471109531
// MI455X (gfx1250) — matrix operand order repaired and checked by reading, not compiled, not run on hardware
//
#include <hip/hip_runtime.h>
#include <hip/hip_bf16.h>

typedef __attribute__((ext_vector_type(16))) _Float16     v16h;
typedef __attribute__((ext_vector_type(8)))  float        v8f;
typedef __attribute__((ext_vector_type(4)))  unsigned int u32x4;
typedef __attribute__((ext_vector_type(8)))  int          i32x8;
typedef __attribute__((ext_vector_type(4)))  int          i32x4;

#define TILE_M 128
#define TILE_N 128
#define TILE_K 32   // halves

// ---------------------------------------------------------------------------
// TDM: issue one tensor_load_to_lds for a 2D tile of 16 dwords x 128 rows
// (= 32 f16 x 128 rows), row stride strideDW dwords, landing contiguously in
// LDS at lds_off. D# built per CDNA5 ISA ch.8:
//   group0: [1:0]=count=1, [63:32]=lds_addr, [120:64]=global_addr, [127:126]=2
//   group1: [17:16]=data_size(2 -> 4B), tensor_dim0=16, tensor_dim1=128,
//           tile_dim0=16, tile_dim1=128, tensor_dim0_stride=strideDW
//   groups 2/3 (+ trailing group): zero (2D tensor, dims 2+ unused)
// This toolchain exposes the 6-arg builtin (g0, g1, g2, g3, g4, cpol).
// ---------------------------------------------------------------------------
__device__ __forceinline__ void tdm_load_tile_16x128(unsigned lds_off,
                                                     const void* gptr,
                                                     int strideDW) {
  unsigned long long ga = (unsigned long long)(size_t)gptr;
  u32x4 g0;
  g0[0] = 1u;                                   // count=1 (valid), no gather
  g0[1] = lds_off;                              // lds_addr (bytes)
  g0[2] = (unsigned)ga;                         // global_addr[31:0]
  g0[3] = ((unsigned)(ga >> 32) & 0x01FFFFFFu)  // global_addr[56:32]
          | 0x80000000u;                        // type=2 ("image")
  i32x8 g1;
  g1[0] = 0x20000;      // workgroup_mask=0, data_size=2 (4B), no pad/iterate
  g1[1] = 16 << 16;     // tensor_dim0[15:0]=16 (dwords)
  g1[2] = 128 << 16;    // tensor_dim0[31:16]=0 | tensor_dim1[15:0]=128
  g1[3] = 16 << 16;     // tensor_dim1[31:16]=0 | tile_dim0=16
  g1[4] = 128;          // tile_dim1=128, tile_dim2=0
  g1[5] = strideDW;     // tensor_dim0_stride[31:0] (dwords)
  g1[6] = 0;            // stride[47:32]=0, tensor_dim1_stride low=0
  g1[7] = 0;
  i32x4 z4;
  z4[0] = 0; z4[1] = 0; z4[2] = 0; z4[3] = 0;
  i32x8 z8;
  z8[0] = 0; z8[1] = 0; z8[2] = 0; z8[3] = 0;
  z8[4] = 0; z8[5] = 0; z8[6] = 0; z8[7] = 0;
  __builtin_amdgcn_tensor_load_to_lds(g0, g1, z4, z4, z8, 0);
}

// ---------------------------------------------------------------------------
// Degree / normalization: deg[i] = 1 (self loop) + #edges with row==i,
// then dinv = rsqrt(deg).
// ---------------------------------------------------------------------------
__global__ void k_deg_init(float* dinv, int N) {
  int i = blockIdx.x * blockDim.x + threadIdx.x;
  if (i < N) dinv[i] = 1.0f;
}

__global__ void k_deg_accum(const int* __restrict__ erow, float* dinv, int E) {
  int e = blockIdx.x * blockDim.x + threadIdx.x;
  if (e < E) atomicAdd(&dinv[erow[e]], 1.0f);
}

__global__ void k_deg_fin(float* dinv, int N) {
  int i = blockIdx.x * blockDim.x + threadIdx.x;
  if (i < N) dinv[i] = rsqrtf(dinv[i]);
}

// ---------------------------------------------------------------------------
// Fused concat [x | patches | doc] -> f16, padded to Kpad=1088 cols and Npad
// rows (zeros). One elementwise pass; removes all branching from the GEMM.
// ---------------------------------------------------------------------------
__global__ void k_concat_f16(const float* __restrict__ x,
                             const float* __restrict__ pat,
                             const float* __restrict__ doc,
                             _Float16* __restrict__ H, int N, int total) {
  int i = blockIdx.x * blockDim.x + threadIdx.x;
  if (i >= total) return;
  int row = i / 1088, c = i - row * 1088;
  float v = 0.0f;
  if (row < N) {
    if      (c < 3)    v = x[row * 3 + c];
    else if (c < 1027) v = pat[(size_t)row * 1024 + (c - 3)];
    else if (c < 1060) v = doc[row * 33 + (c - 1027)];
  }
  H[i] = (_Float16)v;
}

// Transposed f16 weights: Wt[n][k] = W[k][n], k padded to Kpad with zeros.
// Makes the B tile identical in shape/stride to the A tile for TDM.
__global__ void k_transpose_w_f16(const float* __restrict__ W,
                                  _Float16* __restrict__ Wt,
                                  int Kreal, int Kpad, int Ncols) {
  int i = blockIdx.x * blockDim.x + threadIdx.x;
  if (i >= Ncols * Kpad) return;
  int n = i / Kpad, k = i - n * Kpad;
  float v = (k < Kreal) ? W[(size_t)k * Ncols + n] : 0.0f;
  Wt[i] = (_Float16)v;
}

// f32 activations -> f16 rows, zero-padded to Npad rows.
__global__ void k_rows_to_f16(const float* __restrict__ src,
                              _Float16* __restrict__ dst, int N, int F,
                              int total) {
  int i = blockIdx.x * blockDim.x + threadIdx.x;
  if (i >= total) return;
  int row = i / F;
  dst[i] = (row < N) ? (_Float16)src[i] : (_Float16)0.0f;
}

// ---------------------------------------------------------------------------
// WMMA GEMM with TDM staging: C[N,Ncols] = A16[Npad,Kh] @ Bt16[Ncols,Kh]^T.
// Block = 256 threads = 8 waves, 128x128 C tile; wave0 issues two
// tensor_load_to_lds per K-step (A tile m-major, B tile n-major), waits
// TENSORcnt, block barriers, then all waves run 2x4 v_wmma_f32_16x16x32_f16.
// Fragment LDS reads are contiguous b128s per the CDNA5 16-bit A/B layouts.
// ---------------------------------------------------------------------------
__global__ __launch_bounds__(256)
void k_gemm_wmma_tdm(const _Float16* __restrict__ A16,
                     const _Float16* __restrict__ Bt16,
                     float* __restrict__ C,
                     int Kh, int N, int Ncols) {
  __shared__ _Float16 As[TILE_M][TILE_K];   // [m][k]
  __shared__ _Float16 Bs[TILE_N][TILE_K];   // [n][k]

  const int tid  = threadIdx.x;
  const int lane = tid & 31;
  const int wave = tid >> 5;
  const int rowBase = blockIdx.x * TILE_M;
  const int colBase = blockIdx.y * TILE_N;
  const int rowBlock = (wave >> 1) * 32;  // 0,32,64,96
  const int colBlock = (wave & 1) * 64;   // 0,64

  v8f acc[2][4];
  #pragma unroll
  for (int rb = 0; rb < 2; ++rb)
    #pragma unroll
    for (int cb = 0; cb < 4; ++cb)
      acc[rb][cb] = (v8f){};

  const unsigned ldsA = (unsigned)(size_t)&As[0][0];  // low 32 bits of flat
  const unsigned ldsB = (unsigned)(size_t)&Bs[0][0];  // addr = LDS offset
  const int strideDW = Kh >> 1;
  const _Float16* aTile = A16 + (size_t)rowBase * Kh;
  const _Float16* bTile = Bt16 + (size_t)colBase * Kh;

  const int numK = Kh / TILE_K;
  for (int kt = 0; kt < numK; ++kt) {
    if (tid < 32) {  // wave-uniform: only wave 0 drives the DMA
      tdm_load_tile_16x128(ldsA, aTile + kt * TILE_K, strideDW);
      tdm_load_tile_16x128(ldsB, bTile + kt * TILE_K, strideDW);
      __builtin_amdgcn_s_wait_tensorcnt(0);
    }
    __syncthreads();  // tiles visible to all waves

    // A fragment (16x32 f16): lane m=l&15; lanes<16 hold K {0..7,16..23},
    // lanes>=16 hold K {8..15,24..31}.
    const int kbaseA = (lane >= 16) ? 8 : 0;
    // B fragment (32x16 f16): lane n=l&15; same K split as A.
    const int koffB  = (lane >= 16) ? 8 : 0;

    v16h afrag[2];
    #pragma unroll
    for (int rb = 0; rb < 2; ++rb) {
      const _Float16* ap = &As[rowBlock + rb * 16 + (lane & 15)][0];
      #pragma unroll
      for (int h = 0; h < 8; ++h) {
        afrag[rb][h]     = ap[kbaseA + h];
        afrag[rb][8 + h] = ap[16 + kbaseA + h];
      }
    }
    v16h bfrag[4];
    #pragma unroll
    for (int cb = 0; cb < 4; ++cb) {
      const _Float16* bp = &Bs[colBlock + cb * 16 + (lane & 15)][0];
      #pragma unroll
      for (int h = 0; h < 8; ++h) {
        bfrag[cb][h]     = bp[koffB + h];
        bfrag[cb][8 + h] = bp[16 + koffB + h];
      }
    }

    #pragma unroll
    for (int rb = 0; rb < 2; ++rb)
      #pragma unroll
      for (int cb = 0; cb < 4; ++cb)
        acc[rb][cb] = __builtin_amdgcn_wmma_f32_16x16x32_f16(
            false, afrag[rb], false, bfrag[cb],
            (short)0, acc[rb][cb], false, false);
    __syncthreads();  // all reads done before next DMA overwrites tiles
  }

  // Epilogue: C/D layout — VGPR r holds row r (+8 for lanes>=16), col = l&15.
  const int mOff = (lane >= 16) ? 8 : 0;
  const int nIdx = lane & 15;
  #pragma unroll
  for (int rb = 0; rb < 2; ++rb)
    #pragma unroll
    for (int cb = 0; cb < 4; ++cb)
      #pragma unroll
      for (int r = 0; r < 8; ++r) {
        int grow = rowBase + rowBlock + rb * 16 + r + mOff;
        int gcol = colBase + colBlock + cb * 16 + nIdx;
        if (grow < N) C[(size_t)grow * Ncols + gcol] = acc[rb][cb][r];
      }
}

// ---------------------------------------------------------------------------
// out[i][f] = b[f] + dinv[i]^2 * XW[i][f]   (self-loop term + bias), float4.
// ---------------------------------------------------------------------------
__global__ void k_init_out(const float4* __restrict__ XW,
                           const float4* __restrict__ b,
                           const float* __restrict__ dinv,
                           float4* __restrict__ out, int N, int Fq) {
  int i = blockIdx.x * blockDim.x + threadIdx.x;
  if (i >= N * Fq) return;
  int node = i / Fq, j = i % Fq;
  float d = dinv[node];
  float w = d * d;
  float4 v = XW[i];
  float4 bb = b[j];
  out[i] = make_float4(bb.x + w * v.x, bb.y + w * v.y,
                       bb.z + w * v.z, bb.w + w * v.w);
}

// ---------------------------------------------------------------------------
// Edge scatter: out[row[e]] += dinv[row]*dinv[col] * XW[col[e]], float4
// chunks, fp32 global atomics.
// ---------------------------------------------------------------------------
__global__ void k_edge_scatter(const float* __restrict__ XW,
                               float* __restrict__ out,
                               const int* __restrict__ erow,
                               const int* __restrict__ ecol,
                               const float* __restrict__ dinv,
                               int E, int F) {
  int tid = blockIdx.x * blockDim.x + threadIdx.x;
  int perEdge = F >> 2;
  int e = tid / perEdge;
  if (e >= E) return;
  int j = (tid - e * perEdge) << 2;
  int r = erow[e], c = ecol[e];
  float w = dinv[r] * dinv[c];
  const float4 v = *reinterpret_cast<const float4*>(XW + (size_t)c * F + j);
  float* dst = out + (size_t)r * F + j;
  atomicAdd(dst + 0, w * v.x);
  atomicAdd(dst + 1, w * v.y);
  atomicAdd(dst + 2, w * v.z);
  atomicAdd(dst + 3, w * v.w);
}

// ---------------------------------------------------------------------------
// Final projection [N,896]@[896,18] + bias + softmax. One wave32 per node;
// row staged in LDS, lanes 0..17 own one class each; wave32 shuffle softmax.
// ---------------------------------------------------------------------------
__global__ __launch_bounds__(256)
void k_proj_softmax(const float* __restrict__ out1,
                    const float* __restrict__ out2,
                    const float* __restrict__ out3,
                    const float* __restrict__ Wp,
                    const float* __restrict__ bp,
                    float* __restrict__ out, int N) {
  __shared__ float sh[8][896];
  int wave = threadIdx.x >> 5;
  int lane = threadIdx.x & 31;
  int node = blockIdx.x * 8 + wave;
  if (node >= N) return;  // wave-uniform exit; no block barriers below

  for (int k = lane; k < 512; k += 32) sh[wave][k]       = out1[(size_t)node * 512 + k];
  for (int k = lane; k < 256; k += 32) sh[wave][512 + k] = out2[(size_t)node * 256 + k];
  for (int k = lane; k < 128; k += 32) sh[wave][768 + k] = out3[(size_t)node * 128 + k];
  asm volatile("s_wait_dscnt 0" ::: "memory");  // per-wave LDS RAW fence

  float accv = 0.0f;
  if (lane < 18) {
    accv = bp[lane];
    for (int k = 0; k < 896; ++k) accv += sh[wave][k] * Wp[k * 18 + lane];
  }
  float m = (lane < 18) ? accv : -3.4e38f;
  #pragma unroll
  for (int off = 16; off >= 1; off >>= 1) m = fmaxf(m, __shfl_xor(m, off, 32));
  float ex = (lane < 18) ? __expf(accv - m) : 0.0f;
  float s = ex;
  #pragma unroll
  for (int off = 16; off >= 1; off >>= 1) s += __shfl_xor(s, off, 32);
  if (lane < 18) out[(size_t)node * 18 + lane] = ex / s;
}

// ---------------------------------------------------------------------------
// Host launcher
// ---------------------------------------------------------------------------
extern "C" void kernel_launch(void* const* d_in, const int* in_sizes, int n_in,
                              void* d_out, int out_size, void* d_ws, size_t ws_size,
                              hipStream_t stream) {
  const float* x   = (const float*)d_in[0];
  const float* pat = (const float*)d_in[1];
  const float* doc = (const float*)d_in[2];
  const int*  eidx = (const int*)  d_in[3];
  const float* W1  = (const float*)d_in[4];
  const float* b1  = (const float*)d_in[5];
  const float* W2  = (const float*)d_in[6];
  const float* b2  = (const float*)d_in[7];
  const float* W3  = (const float*)d_in[8];
  const float* b3  = (const float*)d_in[9];
  const float* Wp  = (const float*)d_in[10];
  const float* bp  = (const float*)d_in[11];

  const int N = in_sizes[0] / 3;
  const int E = in_sizes[3] / 2;
  const int* erow = eidx;
  const int* ecol = eidx + E;

  const int T = 256;
  const int rowTiles = (N + TILE_M - 1) / TILE_M;
  const int Npad = rowTiles * TILE_M;  // padded row count for TDM tiles

  // Workspace layout (256B-aligned segments)
  float* ws   = (float*)d_ws;
  size_t nAl  = ((size_t)N + 63) & ~(size_t)63;
  float* dinv = ws;
  float* tmp  = dinv + nAl;                 // reusable GEMM output [N,512] f32
  float* out1 = tmp  + (size_t)N * 512;
  float* out2 = out1 + (size_t)N * 512;
  float* out3 = out2 + (size_t)N * 256;
  _Float16* Hf16  = (_Float16*)(out3 + (size_t)N * 128);  // [Npad,1088]
  _Float16* W1t   = Hf16  + (size_t)Npad * 1088;          // [512,1088]
  _Float16* W2t   = W1t   + (size_t)512 * 1088;           // [256,512]
  _Float16* W3t   = W2t   + (size_t)256 * 512;            // [128,256]
  _Float16* out1h = W3t   + (size_t)128 * 256;            // [Npad,512]
  _Float16* out2h = out1h + (size_t)Npad * 512;           // [Npad,256]
  (void)out_size; (void)ws_size; (void)n_in;

  // 1) normalization
  k_deg_init <<<(N + T - 1) / T, T, 0, stream>>>(dinv, N);
  k_deg_accum<<<(E + T - 1) / T, T, 0, stream>>>(erow, dinv, E);
  k_deg_fin  <<<(N + T - 1) / T, T, 0, stream>>>(dinv, N);

  // 2) f16 operand preparation (concat fused, weights transposed+padded)
  {
    int tot = Npad * 1088;
    k_concat_f16<<<(tot + T - 1) / T, T, 0, stream>>>(x, pat, doc, Hf16, N, tot);
  }
  k_transpose_w_f16<<<(512 * 1088 + T - 1) / T, T, 0, stream>>>(W1, W1t, 1060, 1088, 512);
  k_transpose_w_f16<<<(256 * 512  + T - 1) / T, T, 0, stream>>>(W2, W2t, 512, 512, 256);
  k_transpose_w_f16<<<(128 * 256  + T - 1) / T, T, 0, stream>>>(W3, W3t, 256, 256, 128);

  // 3) Layer 1: TDM+WMMA GEMM, then init + edge scatter
  k_gemm_wmma_tdm<<<dim3(rowTiles, 4), T, 0, stream>>>(Hf16, W1t, tmp, 1088, N, 512);
  k_init_out<<<(N * 128 + T - 1) / T, T, 0, stream>>>(
      (const float4*)tmp, (const float4*)b1, dinv, (float4*)out1, N, 128);
  k_edge_scatter<<<(E * 128 + T - 1) / T, T, 0, stream>>>(tmp, out1, erow, ecol, dinv, E, 512);

  // 4) Layer 2
  {
    int tot = Npad * 512;
    k_rows_to_f16<<<(tot + T - 1) / T, T, 0, stream>>>(out1, out1h, N, 512, tot);
  }
  k_gemm_wmma_tdm<<<dim3(rowTiles, 2), T, 0, stream>>>(out1h, W2t, tmp, 512, N, 256);
  k_init_out<<<(N * 64 + T - 1) / T, T, 0, stream>>>(
      (const float4*)tmp, (const float4*)b2, dinv, (float4*)out2, N, 64);
  k_edge_scatter<<<(E * 64 + T - 1) / T, T, 0, stream>>>(tmp, out2, erow, ecol, dinv, E, 256);

  // 5) Layer 3
  {
    int tot = Npad * 256;
    k_rows_to_f16<<<(tot + T - 1) / T, T, 0, stream>>>(out2, out2h, N, 256, tot);
  }
  k_gemm_wmma_tdm<<<dim3(rowTiles, 1), T, 0, stream>>>(out2h, W3t, tmp, 256, N, 128);
  k_init_out<<<(N * 32 + T - 1) / T, T, 0, stream>>>(
      (const float4*)tmp, (const float4*)b3, dinv, (float4*)out3, N, 32);
  k_edge_scatter<<<(E * 32 + T - 1) / T, T, 0, stream>>>(tmp, out3, erow, ecol, dinv, E, 128);

  // 6) projection + softmax -> d_out [N,18]
  k_proj_softmax<<<(N + 7) / 8, T, 0, stream>>>(out1, out2, out3, Wp, bp, (float*)d_out, N);
}
